// ModalMoE_53412213293357
// MI455X (gfx1250) — matrix operand order repaired and checked by reading, not compiled, not run on hardware
//
#include <hip/hip_runtime.h>
#include <stdint.h>

// ---- CDNA5 WMMA vector types --------------------------------------------
typedef __bf16 v16bf __attribute__((ext_vector_type(16)));
typedef float  v8f   __attribute__((ext_vector_type(8)));

#define BATCH 4096
#define DTOT  2560            // 768 + 1024 + 512 + 256
#define NOUT  2048
#define NK    (DTOT / 32)     // 80 K-tiles
#define LDS_STRIDE 20         // dwords per 32-bf16 row (16 dw data + 4 dw pad)
#define ABOFF (256 * LDS_STRIDE)           // B region after 256 A rows
#define BUFSZ (256 * LDS_STRIDE + 128 * LDS_STRIDE)   // 7680 dwords / buffer

struct Frag32 { uint4 lo, hi; };   // 32 bytes == one v16bf fragment

// ---- gfx1250 async global->LDS copy (ASYNCcnt path), with safe fallback --
#if __has_builtin(__builtin_amdgcn_global_load_async_to_lds_b128)
#define HAVE_ASYNC_LDS 1
#else
#define HAVE_ASYNC_LDS 0
#endif

// Builtin prototype (from hipcc diagnostic): param0 is
// '__attribute__((__vector_size__(4*sizeof(int)))) int __device__ *'
// i.e. int4 in address space 1; param1 is the LDS-side int4 pointer.
typedef int v4i __attribute__((vector_size(16)));
typedef __attribute__((address_space(1))) v4i* gas_b128_t;
typedef __attribute__((address_space(3))) v4i* las_b128_t;

__device__ __forceinline__ void copy_b128_to_lds(const unsigned short* g, uint32_t* l) {
#if HAVE_ASYNC_LDS
    __builtin_amdgcn_global_load_async_to_lds_b128(
        (gas_b128_t)(uintptr_t)g,
        (las_b128_t)l, 0, 0);
#else
    *(uint4*)l = *(const uint4*)g;
#endif
}

__device__ __forceinline__ void wait_staged() {
#if HAVE_ASYNC_LDS
 #if __has_builtin(__builtin_amdgcn_s_wait_asynccnt)
    __builtin_amdgcn_s_wait_asynccnt(0);
 #else
    asm volatile("s_wait_asynccnt 0" ::: "memory");
 #endif
#endif
    __syncthreads();
}

__device__ __forceinline__ unsigned short f2bf(float x) {
    union { float f; uint32_t u; } v; v.f = x;
    uint32_t u = v.u;
    uint32_t r = u + 0x7FFFu + ((u >> 16) & 1u);   // round-to-nearest-even
    return (unsigned short)(r >> 16);
}

// ---- Kernel 1: gating + gate-scaled bf16 pack of A' ----------------------
// One wave32 per batch row. 8 waves / block.
__global__ __launch_bounds__(256) void gate_pack_kernel(
    const float* __restrict__ f0, const float* __restrict__ f1,
    const float* __restrict__ f2, const float* __restrict__ f3,
    const float* __restrict__ gw, const float* __restrict__ gb,
    unsigned short* __restrict__ Ab, float* __restrict__ gates)
{
    const int wave = threadIdx.x >> 5;
    const int lane = threadIdx.x & 31;
    const int row  = blockIdx.x * 8 + wave;

    const float* fp[4] = { f0 + (size_t)row * 768, f1 + (size_t)row * 1024,
                           f2 + (size_t)row * 512, f3 + (size_t)row * 256 };
    const int dims[4] = { 768, 1024, 512, 256 };
    const int offs[4] = { 0, 768, 1792, 2304 };

    float s[4] = { 0.f, 0.f, 0.f, 0.f };
    #pragma unroll
    for (int i = 0; i < 4; ++i) {
        for (int k = lane; k < dims[i]; k += 32) {
            float x = fp[i][k];
            s[0] += x * gw[0 * DTOT + offs[i] + k];
            s[1] += x * gw[1 * DTOT + offs[i] + k];
            s[2] += x * gw[2 * DTOT + offs[i] + k];
            s[3] += x * gw[3 * DTOT + offs[i] + k];
        }
    }
    #pragma unroll
    for (int m = 16; m >= 1; m >>= 1) {
        #pragma unroll
        for (int e = 0; e < 4; ++e) s[e] += __shfl_xor(s[e], m, 32);
    }
    #pragma unroll
    for (int e = 0; e < 4; ++e) s[e] += gb[e];

    float mx = fmaxf(fmaxf(s[0], s[1]), fmaxf(s[2], s[3]));
    float p[4], den = 0.f;
    #pragma unroll
    for (int e = 0; e < 4; ++e) { p[e] = __expf(s[e] - mx); den += p[e]; }
    float inv = 1.0f / den;
    #pragma unroll
    for (int e = 0; e < 4; ++e) p[e] *= inv;

    int i1 = 0;
    #pragma unroll
    for (int e = 1; e < 4; ++e) if (p[e] > p[i1]) i1 = e;
    int i2 = -1;
    #pragma unroll
    for (int e = 0; e < 4; ++e) {
        if (e == i1) continue;
        if (i2 < 0 || p[e] > p[i2]) i2 = e;
    }
    float g[4];
    #pragma unroll
    for (int e = 0; e < 4; ++e) g[e] = (e == i1 || e == i2) ? p[e] : 0.0f;

    if (lane < 4) gates[(size_t)row * 4 + lane] = g[lane];

    #pragma unroll
    for (int i = 0; i < 4; ++i) {
        float gi = g[i];
        unsigned short* dst = Ab + (size_t)row * DTOT + offs[i];
        for (int k = lane; k < dims[i]; k += 32)
            dst[k] = f2bf(gi * fp[i][k]);
    }
}

// ---- Kernel 2: pack W_cat[f, 0:2560] = concat_e(w_e[f,:]) as bf16 --------
__global__ __launch_bounds__(256) void wpack_kernel(
    const float* __restrict__ w0, const float* __restrict__ w1,
    const float* __restrict__ w2, const float* __restrict__ w3,
    unsigned short* __restrict__ Wb)
{
    int idx = blockIdx.x * 256 + threadIdx.x;   // < NOUT * DTOT
    int f = idx / DTOT, k = idx % DTOT;
    float v;
    if      (k <  768) v = w0[(size_t)f *  768 + k];
    else if (k < 1792) v = w1[(size_t)f * 1024 + (k -  768)];
    else if (k < 2304) v = w2[(size_t)f *  512 + (k - 1792)];
    else               v = w3[(size_t)f *  256 + (k - 2304)];
    Wb[(size_t)f * DTOT + k] = f2bf(v);
}

// ---- staging: one 256x32 A tile + 128x32 B tile into LDS buffer ----------
__device__ __forceinline__ void stage_tile(
    const unsigned short* __restrict__ Ab, const unsigned short* __restrict__ Wb,
    uint32_t* lds, int buf, int mBase, int nBase, int k0, int tid)
{
    uint32_t* base = lds + buf * BUFSZ;
    #pragma unroll
    for (int u = 0; u < 4; ++u) {              // A: 256 rows x 4 b128 quads
        int unit = tid + u * 256;
        int row = unit >> 2, q = unit & 3;
        copy_b128_to_lds(Ab + (size_t)(mBase + row) * DTOT + k0 + q * 8,
                         base + row * LDS_STRIDE + q * 4);
    }
    #pragma unroll
    for (int u = 0; u < 2; ++u) {              // B: 128 rows x 4 b128 quads
        int unit = tid + u * 256;
        int row = unit >> 2, q = unit & 3;
        copy_b128_to_lds(Wb + (size_t)(nBase + row) * DTOT + k0 + q * 8,
                         base + ABOFF + row * LDS_STRIDE + q * 4);
    }
}

// ---- Kernel 3: fused MoE GEMM, out = A'(bf16) x Wcat^T(bf16) + g.b -------
// 256x128 block tile, 8 waves (4M x 2N), each wave 64x64 = 4x4 WMMA tiles.
// Double-buffered LDS (60 KB), one barrier per K-step.
__global__ __launch_bounds__(256) void moe_gemm_kernel(
    const unsigned short* __restrict__ Ab,   // [4096][2560] bf16
    const unsigned short* __restrict__ Wb,   // [2048][2560] bf16
    const float* __restrict__ gates,         // [4096][4]
    const float* __restrict__ b0, const float* __restrict__ b1,
    const float* __restrict__ b2, const float* __restrict__ b3,
    float* __restrict__ out)                 // [4096][2048]
{
    __shared__ uint32_t lds[2 * BUFSZ];      // 61440 B

    const int tid   = threadIdx.x;
    const int lane  = tid & 31;
    const int wave  = tid >> 5;
    const int mBase = blockIdx.y * 256;
    const int nBase = blockIdx.x * 128;
    const int mGrp  = wave >> 1;        // 0..3 -> 64-row slab
    const int nGrp  = wave & 1;         // 0..1 -> 64-col slab
    const int half  = lane >> 4;
    const int lrow  = lane & 15;

    v8f acc[4][4] = {};

    stage_tile(Ab, Wb, lds, 0, mBase, nBase, 0, tid);
    wait_staged();

    for (int kt = 0; kt < NK; ++kt) {
        const int cur = kt & 1;
        if (kt + 1 < NK)
            stage_tile(Ab, Wb, lds, cur ^ 1, mBase, nBase, (kt + 1) * 32, tid);

        const uint32_t* Al = lds + cur * BUFSZ;
        const uint32_t* Bl = Al + ABOFF;

        // B fragments first (held in VGPRs; K = 16h..16h+15 contiguous)
        v16bf bfr[4];
        #pragma unroll
        for (int ni = 0; ni < 4; ++ni) {
            int brow = nGrp * 64 + ni * 16 + lrow;
            Frag32 t;
            t.lo = *(const uint4*)&Bl[brow * LDS_STRIDE + half * 8];
            t.hi = *(const uint4*)&Bl[brow * LDS_STRIDE + half * 8 + 4];
            bfr[ni] = __builtin_bit_cast(v16bf, t);
        }
        // A fragments (K = 16h..16h+15 contiguous) + 16 WMMAs
        #pragma unroll
        for (int mi = 0; mi < 4; ++mi) {
            int arow = mGrp * 64 + mi * 16 + lrow;
            Frag32 t;
            t.lo = *(const uint4*)&Al[arow * LDS_STRIDE + half * 8];
            t.hi = *(const uint4*)&Al[arow * LDS_STRIDE + half * 8 + 4];
            v16bf afr = __builtin_bit_cast(v16bf, t);
            #pragma unroll
            for (int ni = 0; ni < 4; ++ni)
                acc[mi][ni] = __builtin_amdgcn_wmma_f32_16x16x32_bf16(
                    false, afr, false, bfr[ni], (short)0, acc[mi][ni],
                    false, false);
        }
        wait_staged();   // async done + barrier: next buffer ready for all
    }

    // ---- epilogue: out = acc + sum_e g[b,e] * b_e[n] ---------------------
    const float4* g4 = (const float4*)gates;
    float4 bv[4];
    #pragma unroll
    for (int ni = 0; ni < 4; ++ni) {
        int n = nBase + nGrp * 64 + ni * 16 + lrow;
        bv[ni] = make_float4(b0[n], b1[n], b2[n], b3[n]);
    }
    #pragma unroll
    for (int mi = 0; mi < 4; ++mi) {
        #pragma unroll
        for (int j = 0; j < 8; ++j) {
            int m = mBase + mGrp * 64 + mi * 16 + half * 8 + j;  // C/D row map
            float4 g = g4[m];
            #pragma unroll
            for (int ni = 0; ni < 4; ++ni) {
                int n = nBase + nGrp * 64 + ni * 16 + lrow;
                float gb = g.x * bv[ni].x + g.y * bv[ni].y +
                           g.z * bv[ni].z + g.w * bv[ni].w;
                out[(size_t)m * NOUT + n] = acc[mi][ni][j] + gb;
            }
        }
    }
}

extern "C" void kernel_launch(void* const* d_in, const int* in_sizes, int n_in,
                              void* d_out, int out_size, void* d_ws, size_t ws_size,
                              hipStream_t stream) {
    const float* f0 = (const float*)d_in[0];
    const float* f1 = (const float*)d_in[1];
    const float* f2 = (const float*)d_in[2];
    const float* f3 = (const float*)d_in[3];
    const float* gw = (const float*)d_in[4];
    const float* gb = (const float*)d_in[5];
    const float* w0 = (const float*)d_in[6];
    const float* b0 = (const float*)d_in[7];
    const float* w1 = (const float*)d_in[8];
    const float* b1 = (const float*)d_in[9];
    const float* w2 = (const float*)d_in[10];
    const float* b2 = (const float*)d_in[11];
    const float* w3 = (const float*)d_in[12];
    const float* b3 = (const float*)d_in[13];

    // workspace: A' bf16 (20 MB) | Wcat bf16 (10 MB) | gates f32 (64 KB)
    unsigned short* Ab = (unsigned short*)d_ws;
    unsigned short* Wb = Ab + (size_t)BATCH * DTOT;
    float* gates = (float*)(Wb + (size_t)NOUT * DTOT);

    gate_pack_kernel<<<BATCH / 8, 256, 0, stream>>>(f0, f1, f2, f3, gw, gb, Ab, gates);
    wpack_kernel<<<(NOUT * DTOT) / 256, 256, 0, stream>>>(w0, w1, w2, w3, Wb);
    moe_gemm_kernel<<<dim3(NOUT / 128, BATCH / 256), 256, 0, stream>>>(
        Ab, Wb, gates, b0, b1, b2, b3, (float*)d_out);
}
